// SatEncoder_24507083391233
// MI455X (gfx1250) — matrix operand order repaired and checked by reading, not compiled, not run on hardware
//
#include <hip/hip_runtime.h>
#include <hip/hip_bf16.h>

// ---------------------------------------------------------------------------
// SAT encoder for MI455X (gfx1250, wave32, WMMA).
// Heavy work = streaming two 16384x4096 f32 matrices through
// v_wmma_f32_16x16x32_bf16 with on-the-fly f32->bf16 conversion.
// ---------------------------------------------------------------------------

typedef __bf16 bf16;
typedef __attribute__((ext_vector_type(2)))  __bf16 bf16x2;
typedef __attribute__((ext_vector_type(8)))  __bf16 bf16x8;
typedef __attribute__((ext_vector_type(16))) __bf16 bf16x16;
typedef __attribute__((ext_vector_type(8)))  float  f32x8;
typedef __attribute__((ext_vector_type(4)))  float  f32x4;

#define N_CL    16384   // clauses (C)
#define N_V     4096    // variables (V)
#define DVF     104     // final variable feature width
#define NKCH    8       // split-K chunks for gemm_bwd
#define KCHUNK  (N_CL / NKCH)

union ABf {
    bf16x16 v;
    bf16x8  h[2];
    bf16    e[16];
};

__device__ __forceinline__ void cvt8(ABf& A, int base, f32x4 a, f32x4 b) {
    #pragma unroll
    for (int i = 0; i < 4; ++i) {
        A.e[base + i]     = (bf16)a[i];
        A.e[base + 4 + i] = (bf16)b[i];
    }
}

// ---------------------------------------------------------------------------
// Pack vlabels (8 cols, zero-padded to 16) into the WMMA bf16 B layout:
//   pk[ktile*512 + (n + 16*(klocal>=16))*16 + (klocal&15)], ktile = k/32.
// ---------------------------------------------------------------------------
__global__ __launch_bounds__(256) void pack_vlab(const float* __restrict__ vl,
                                                 bf16* __restrict__ pk) {
    int v = blockIdx.x * 256 + threadIdx.x;            // 0..4095
    int khalf = ((v & 31) >= 16) ? 16 : 0;
    int kid = v & 15;
    size_t kt = (size_t)(v >> 5) * 512;
    #pragma unroll
    for (int n = 0; n < 16; ++n) {
        float val = (n < 8) ? vl[(size_t)v * 8 + n] : 0.f;
        pk[kt + (size_t)(n + khalf) * 16 + kid] = (bf16)val;
    }
}

// ---------------------------------------------------------------------------
// gemm_fwd: av[:, col0:col0+dcols] = P @ Xp_new + N @ Xn_new
//   A tiles streamed from f32 cmat rows (contiguous k: lane half hi holds k = 16*hi + 0..15),
//   B tiles read pre-packed bf16 panels (one 16-col panel per ntile).
// block = 64 (dcols<=16) or 128 (dcols=32); grid.x = C/32.
// ---------------------------------------------------------------------------
__global__ __launch_bounds__(128) void gemm_fwd(
        const float* __restrict__ P, const float* __restrict__ Nm,
        const bf16* __restrict__ pkP0, const bf16* __restrict__ pkP1,
        const bf16* __restrict__ pkN0, const bf16* __restrict__ pkN1,
        float* __restrict__ av, int col0, int dcols) {
    const int tid  = threadIdx.x;
    const int lane = tid & 31;
    const int w    = tid >> 5;
    const int mt   = w & 1;          // 2 row tiles of 16 clauses
    const int nt   = w >> 1;         // 0 or 1 column tile of 16
    const int hi   = lane >> 4;
    const int lm   = lane & 15;

    const bf16* pkP = nt ? pkP1 : pkP0;
    const bf16* pkN = nt ? pkN1 : pkN0;

    const size_t row = (size_t)blockIdx.x * 32 + (size_t)mt * 16 + lm;
    const float* pA = P  + row * N_V + hi * 16;  // lane's K sub-chunk base
    const float* nA = Nm + row * N_V + hi * 16;

    f32x8 acc = {0.f, 0.f, 0.f, 0.f, 0.f, 0.f, 0.f, 0.f};

    for (int k = 0; k < N_V; k += 32) {
        __builtin_prefetch((const void*)(pA + k + 256), 0, 0);
        __builtin_prefetch((const void*)(nA + k + 256), 0, 0);

        f32x4 p0 = *(const f32x4*)(pA + k);
        f32x4 p1 = *(const f32x4*)(pA + k + 4);
        f32x4 p2 = *(const f32x4*)(pA + k + 8);
        f32x4 p3 = *(const f32x4*)(pA + k + 12);
        f32x4 q0 = *(const f32x4*)(nA + k);
        f32x4 q1 = *(const f32x4*)(nA + k + 4);
        f32x4 q2 = *(const f32x4*)(nA + k + 8);
        f32x4 q3 = *(const f32x4*)(nA + k + 12);

        ABf aP, aN;
        cvt8(aP, 0, p0, p1); cvt8(aP, 8, p2, p3);
        cvt8(aN, 0, q0, q1); cvt8(aN, 8, q2, q3);

        const size_t bo = (size_t)(k >> 5) * 512 + (size_t)lane * 16;
        bf16x16 bP = *(const bf16x16*)(pkP + bo);
        bf16x16 bN = *(const bf16x16*)(pkN + bo);

        acc = __builtin_amdgcn_wmma_f32_16x16x32_bf16(
                  false, aP.v, false, bP, (short)0, acc, false, false);
        acc = __builtin_amdgcn_wmma_f32_16x16x32_bf16(
                  false, aN.v, false, bN, (short)0, acc, false, false);
    }

    const int n = lm;
    if (nt * 16 + n < dcols) {
        #pragma unroll
        for (int r = 0; r < 8; ++r) {
            size_t orow = (size_t)blockIdx.x * 32 + (size_t)mt * 16 + r + 8 * hi;
            av[orow * DVF + col0 + nt * 16 + n] = acc[r];
        }
    }
}

// ---------------------------------------------------------------------------
// clause_update: s = relu(av[:,0:dv] @ W^T + B); ct = [clabels | s | 0pad]
// writes ct directly as a bf16 B-panel pack (2 ntiles of 16 cols).
// ---------------------------------------------------------------------------
__global__ __launch_bounds__(256) void clause_update(
        const float* __restrict__ av, int dv,
        const float* __restrict__ W, const float* __restrict__ B,
        const float* __restrict__ clabels, bf16* __restrict__ pkCT) {
    __shared__ float sW[16 * DVF];
    __shared__ float sB[16];
    int tid = threadIdx.x;
    for (int i = tid; i < 16 * dv; i += 256) sW[i] = W[i];
    if (tid < 16) sB[tid] = B[tid];
    __syncthreads();

    size_t c = (size_t)blockIdx.x * 256 + tid;
    const float* row = av + c * DVF;

    float s[16];
    #pragma unroll
    for (int j = 0; j < 16; ++j) s[j] = sB[j];
    for (int i = 0; i < dv; ++i) {
        float xi = row[i];
        #pragma unroll
        for (int j = 0; j < 16; ++j) s[j] += sW[j * dv + i] * xi;
    }

    int khalf = ((c & 31) >= 16) ? 16 : 0;
    int kid = (int)(c & 15);
    size_t kt = (size_t)(c >> 5) * 1024;
    #pragma unroll
    for (int j2 = 0; j2 < 32; ++j2) {
        float val = (j2 < 8) ? clabels[c * 8 + j2]
                  : (j2 < 24 ? fmaxf(s[j2 - 8], 0.f) : 0.f);
        pkCT[kt + (size_t)(j2 >> 4) * 512 + (size_t)((j2 & 15) + khalf) * 16 + kid]
            = (bf16)val;
    }
}

// ---------------------------------------------------------------------------
// gemm_bwd: partial[y] = cmat[kchunk]^T @ ct  -> [V, 24] per K chunk.
// Stages 64x64 f32 blocks of cmat with non-temporal loads (single use per
// pass), writes transposed bf16 into LDS, then ds_load_b128 yields WMMA A
// tiles of cmat^T directly.
// grid = (V/64, NKCH), block = 256 (8 waves: 4 mtiles x 2 ntiles).
// ---------------------------------------------------------------------------
__global__ __launch_bounds__(256) void gemm_bwd(
        const float* __restrict__ M, const bf16* __restrict__ pkCT,
        float* __restrict__ outr) {
    __shared__ bf16 T[64 * 64];      // T[v_local][c_local], 8 KB

    const int tid  = threadIdx.x;
    const int lane = tid & 31;
    const int w    = tid >> 5;
    const int mt   = w & 3;
    const int nt   = w >> 2;
    const int hi   = lane >> 4;
    const int lm   = lane & 15;
    const size_t v0 = (size_t)blockIdx.x * 64;
    const int k0 = blockIdx.y * KCHUNK;

    f32x8 acc = {0.f, 0.f, 0.f, 0.f, 0.f, 0.f, 0.f, 0.f};

    for (int k = k0; k < k0 + KCHUNK; k += 64) {
        // stage: 64 c-rows x 64 v-cols, transposed into LDS as bf16
        #pragma unroll
        for (int u = 0; u < 2; ++u) {
            int unit = tid + 256 * u;        // 512 units
            int cp = unit >> 4;              // 0..31 (c pair)
            int vg = unit & 15;              // 0..15 (v group of 4)
            const float* r0p = M + (size_t)(k + 2 * cp) * N_V + v0 + 4 * vg;
            const float* r1p = r0p + N_V;
            __builtin_prefetch((const void*)(r0p + 64 * N_V), 0, 0);
            f32x4 r0 = __builtin_nontemporal_load((const f32x4*)r0p);
            f32x4 r1 = __builtin_nontemporal_load((const f32x4*)r1p);
            #pragma unroll
            for (int i = 0; i < 4; ++i) {
                bf16x2 pr;
                pr[0] = (bf16)r0[i];
                pr[1] = (bf16)r1[i];
                *(bf16x2*)&T[(4 * vg + i) * 64 + 2 * cp] = pr;
            }
        }
        __syncthreads();

        #pragma unroll
        for (int sub = 0; sub < 2; ++sub) {
            int kb = sub * 32 + hi * 16;
            int vl = mt * 16 + lm;
            ABf A;
            A.h[0] = *(const bf16x8*)&T[vl * 64 + kb];
            A.h[1] = *(const bf16x8*)&T[vl * 64 + kb + 8];
            bf16x16 bmat = *(const bf16x16*)(pkCT
                + (size_t)((k >> 5) + sub) * 1024 + (size_t)nt * 512
                + (size_t)lane * 16);
            acc = __builtin_amdgcn_wmma_f32_16x16x32_bf16(
                      false, A.v, false, bmat, (short)0, acc, false, false);
        }
        __syncthreads();
    }

    int j = nt * 16 + lm;
    if (j < 24) {
        float* o = outr + (size_t)blockIdx.y * (size_t)N_V * 24;
        #pragma unroll
        for (int r = 0; r < 8; ++r)
            o[(v0 + mt * 16 + r + 8 * hi) * 24 + j] = acc[r];
    }
}

// ---------------------------------------------------------------------------
// var_update: reduce NKCH split-K partials, relu(x@W_C^T + B_C),
// LayerNorm(16), emit bf16 B-panels pack_pv / pack_nv for next forward GEMM.
// ---------------------------------------------------------------------------
__global__ __launch_bounds__(256) void var_update(
        const float* __restrict__ pvr, const float* __restrict__ nvr,
        const float* __restrict__ WC, const float* __restrict__ BC,
        const float* __restrict__ g, const float* __restrict__ b,
        bf16* __restrict__ pk_pv, bf16* __restrict__ pk_nv) {
    __shared__ float sW[16 * 24];
    __shared__ float sB[16], sg[16], sb[16];
    int tid = threadIdx.x;
    if (tid < 16 * 24) sW[tid] = WC[tid];
    if (tid < 16) { sB[tid] = BC[tid]; sg[tid] = g[tid]; sb[tid] = b[tid]; }
    __syncthreads();

    int v = blockIdx.x * 256 + tid;
    int khalf = ((v & 31) >= 16) ? 16 : 0;
    int kid = v & 15;
    size_t kt = (size_t)(v >> 5) * 512;

    const float* srcs[2] = { pvr, nvr };
    bf16* dsts[2] = { pk_pv, pk_nv };

    for (int s2 = 0; s2 < 2; ++s2) {
        const float* base = srcs[s2] + (size_t)v * 24;
        float rin[24];
        #pragma unroll
        for (int i = 0; i < 24; ++i) {
            float acc = 0.f;
            #pragma unroll
            for (int p = 0; p < NKCH; ++p)
                acc += base[(size_t)p * (size_t)N_V * 24 + i];
            rin[i] = acc;
        }
        float x[16], m = 0.f;
        #pragma unroll
        for (int j = 0; j < 16; ++j) {
            float s = sB[j];
            #pragma unroll
            for (int i = 0; i < 24; ++i) s += sW[j * 24 + i] * rin[i];
            x[j] = fmaxf(s, 0.f);
            m += x[j];
        }
        m *= (1.f / 16.f);
        float var = 0.f;
        #pragma unroll
        for (int j = 0; j < 16; ++j) { float d = x[j] - m; var += d * d; }
        var *= (1.f / 16.f);
        float inv = rsqrtf(var + 1e-5f);
        bf16* pk = dsts[s2];
        #pragma unroll
        for (int j = 0; j < 16; ++j) {
            float outv = (x[j] - m) * inv * sg[j] + sb[j];
            pk[kt + (size_t)(j + khalf) * 16 + kid] = (bf16)outv;
        }
    }
}

// ---------------------------------------------------------------------------
// final_out: out = [clabels | relu(av @ final_v2c^T + final_bias)]
// ---------------------------------------------------------------------------
__global__ __launch_bounds__(256) void final_out(
        const float* __restrict__ av, const float* __restrict__ FW,
        const float* __restrict__ FB, const float* __restrict__ clabels,
        float* __restrict__ out) {
    __shared__ float sW[16 * DVF];
    __shared__ float sB[16];
    int tid = threadIdx.x;
    for (int i = tid; i < 16 * DVF; i += 256) sW[i] = FW[i];
    if (tid < 16) sB[tid] = FB[tid];
    __syncthreads();

    size_t c = (size_t)blockIdx.x * 256 + tid;
    const float* row = av + c * DVF;
    float* o = out + c * 24;
    #pragma unroll
    for (int j = 0; j < 8; ++j) o[j] = clabels[c * 8 + j];
    #pragma unroll
    for (int j = 0; j < 16; ++j) {
        float s = sB[j];
        for (int i = 0; i < DVF; ++i) s += sW[j * DVF + i] * row[i];
        o[8 + j] = fmaxf(s, 0.f);
    }
}

// ---------------------------------------------------------------------------
// Host orchestration
// ---------------------------------------------------------------------------
extern "C" void kernel_launch(void* const* d_in, const int* in_sizes, int n_in,
                              void* d_out, int out_size, void* d_ws, size_t ws_size,
                              hipStream_t stream) {
    (void)in_sizes; (void)n_in; (void)out_size; (void)ws_size;

    const float* vlabels = (const float*)d_in[0];
    const float* clabels = (const float*)d_in[1];
    const float* P       = (const float*)d_in[2];   // cmat_pos [C,V]
    const float* Nm      = (const float*)d_in[3];   // cmat_neg [C,V]
    const float* WL[3]   = { (const float*)d_in[4], (const float*)d_in[5],
                             (const float*)d_in[6] };
    const float* BL      = (const float*)d_in[7];   // [3,16]
    const float* WC      = (const float*)d_in[8];   // [3,16,24]
    const float* BC      = (const float*)d_in[9];   // [3,16]
    const float* g       = (const float*)d_in[10];  // [3,16]
    const float* b       = (const float*)d_in[11];  // [3,16]
    const float* FW      = (const float*)d_in[12];  // [16,104]
    const float* FB      = (const float*)d_in[13];  // [16]
    float* out           = (float*)d_out;

    char* wsp = (char*)d_ws;
    auto carve = [&](size_t bytes) -> char* {
        char* p = wsp;
        wsp += (bytes + 255) & ~(size_t)255;
        return p;
    };
    float* av   = (float*)carve((size_t)N_CL * DVF * 4);          // 6.8 MB
    float* pvr  = (float*)carve((size_t)NKCH * N_V * 24 * 4);     // split-K partials
    float* nvr  = (float*)carve((size_t)NKCH * N_V * 24 * 4);
    bf16* pk_v  = (bf16*)carve((size_t)(N_V / 32) * 512 * 2);     // 128 KB
    bf16* pk_pv = (bf16*)carve((size_t)(N_V / 32) * 512 * 2);
    bf16* pk_nv = (bf16*)carve((size_t)(N_V / 32) * 512 * 2);
    bf16* pkCT  = (bf16*)carve((size_t)(N_CL / 32) * 1024 * 2);   // 1 MB

    // t=0 forward uses vlabels for both polarities (single 16-col panel).
    pack_vlab<<<N_V / 256, 256, 0, stream>>>(vlabels, pk_v);
    gemm_fwd<<<N_CL / 32, 64, 0, stream>>>(P, Nm, pk_v, pk_v, pk_v, pk_v,
                                           av, 0, 8);

    int dv = 8;
    for (int t = 0; t < 3; ++t) {
        clause_update<<<N_CL / 256, 256, 0, stream>>>(av, dv, WL[t],
                                                      BL + t * 16, clabels, pkCT);
        gemm_bwd<<<dim3(N_V / 64, NKCH), 256, 0, stream>>>(P,  pkCT, pvr);
        gemm_bwd<<<dim3(N_V / 64, NKCH), 256, 0, stream>>>(Nm, pkCT, nvr);
        var_update<<<N_V / 256, 256, 0, stream>>>(pvr, nvr, WC + t * 16 * 24,
                                                  BC + t * 16, g + t * 16,
                                                  b + t * 16, pk_pv, pk_nv);
        // append new columns: pos gets [pv, nv], neg gets [nv, pv]
        gemm_fwd<<<N_CL / 32, 128, 0, stream>>>(P, Nm, pk_pv, pk_nv,
                                                pk_nv, pk_pv, av, 8 + 32 * t, 32);
        dv += 32;
    }

    final_out<<<N_CL / 256, 256, 0, stream>>>(av, FW, FB, clabels, out);
}
